// TransformerLayer_19344532701998
// MI455X (gfx1250) — matrix operand order repaired and checked by reading, not compiled, not run on hardware
//
#include <hip/hip_runtime.h>
#include <hip/hip_bf16.h>
#include <math.h>
#include <stdint.h>

// ---------------- problem constants (from reference) ----------------
#define B_   2
#define S_   2048
#define D_   1024
#define H_   16
#define DH_  64
#define DFF_ 4096
#define M_   (B_*S_)   // 4096 total rows

typedef __attribute__((ext_vector_type(16))) __bf16 bf16x16;
typedef __attribute__((ext_vector_type(8)))  __bf16 bf16x8;
typedef __attribute__((ext_vector_type(8)))  float  f32x8;
typedef __attribute__((ext_vector_type(4)))  int    v4i_t;

// ---------------- CDNA5 async global->LDS copy (ASYNCcnt path) ------------
#if defined(__gfx1250__) && __has_builtin(__builtin_amdgcn_global_load_async_to_lds_b128)
#define HAS_ASYNC_LDS 1
typedef __attribute__((address_space(1))) v4i_t* gas_ptr_t;  // global int4*
typedef __attribute__((address_space(3))) v4i_t* las_ptr_t;  // LDS int4*
#else
#define HAS_ASYNC_LDS 0
#endif

// copy 16 bytes (8 bf16) global -> LDS
__device__ __forceinline__ void cp16(const __bf16* g, __bf16* l) {
#if HAS_ASYNC_LDS
  __builtin_amdgcn_global_load_async_to_lds_b128(
      (gas_ptr_t)(uintptr_t)g, (las_ptr_t)(uint32_t)(uintptr_t)l, 0, 0);
#else
  *(bf16x8*)l = *(const bf16x8*)g;
#endif
}
__device__ __forceinline__ void wait_async_all() {
#if HAS_ASYNC_LDS
#if __has_builtin(__builtin_amdgcn_s_wait_asynccnt)
  __builtin_amdgcn_s_wait_asynccnt(0);
#else
  asm volatile("s_wait_asynccnt 0x0" ::: "memory");
#endif
#endif
}

// D = A(16x32 bf16) * B(32x16 bf16) + C(16x16 f32)
__device__ __forceinline__ f32x8 wmma_bf16(bf16x16 a, bf16x16 b, f32x8 c) {
  return __builtin_amdgcn_wmma_f32_16x16x32_bf16(
      /*neg_a=*/false, a, /*neg_b=*/false, b,
      /*c_mod=*/(short)0, c, /*reuse_a=*/false, /*reuse_b=*/false);
}

// A-fragment (16x32, row-major source, per-lane row = lane&15):
// elems e -> k = half*16 + e  (32 contiguous bytes)
__device__ __forceinline__ bf16x16 ld_afrag(const __bf16* rowp, int hb) {
  bf16x16 a;
  ((bf16x8*)&a)[0] = *(const bf16x8*)(rowp + hb * 16);
  ((bf16x8*)&a)[1] = *(const bf16x8*)(rowp + hb * 16 + 8);
  return a;
}
// B-fragment (32x16, source transposed so k is contiguous per lane col):
// elems e -> k = half*16 + e  (32 contiguous bytes)
__device__ __forceinline__ bf16x16 ld_bfrag(const __bf16* rowp, int hb) {
  bf16x16 b;
  ((bf16x8*)&b)[0] = *(const bf16x8*)(rowp + hb * 16);
  ((bf16x8*)&b)[1] = *(const bf16x8*)(rowp + hb * 16 + 8);
  return b;
}

// ---------------- fused transpose + fp32->bf16 convert --------------------
// in: [z][K][N] f32 row-major ; out: [z][N][K] bf16 row-major
__global__ __launch_bounds__(256)
void cvt_transpose(const float* __restrict__ src, __bf16* __restrict__ dst,
                   int K, int N) {
  __shared__ float t[32][33];
  const int z = blockIdx.z;
  const int kb = blockIdx.y * 32, nb = blockIdx.x * 32;
  const int tx = threadIdx.x & 31, ty = threadIdx.x >> 5;  // ty: 0..7
  const size_t zoff = (size_t)z * K * N;
#pragma unroll
  for (int i = ty; i < 32; i += 8)
    t[i][tx] = src[zoff + (size_t)(kb + i) * N + nb + tx];
  __syncthreads();
#pragma unroll
  for (int i = ty; i < 32; i += 8)
    dst[zoff + (size_t)(nb + i) * K + kb + tx] = (__bf16)t[tx][i];
}

// ---------------- RMSNorm (fp32 in, bf16 out) ----------------
__global__ __launch_bounds__(256)
void rmsnorm_bf16(const float* __restrict__ x, const float* __restrict__ g,
                  __bf16* __restrict__ out) {
  const int row = blockIdx.x;
  const int tid = threadIdx.x;
  const float* xr = x + (size_t)row * D_;
  float ss = 0.f;
#pragma unroll
  for (int i = 0; i < D_ / 256; ++i) { float v = xr[tid + i * 256]; ss += v * v; }
#pragma unroll
  for (int m = 16; m >= 1; m >>= 1) ss += __shfl_xor(ss, m, 32);
  __shared__ float red[8];
  if ((tid & 31) == 0) red[tid >> 5] = ss;
  __syncthreads();
  float tot = 0.f;
#pragma unroll
  for (int wv = 0; wv < 8; ++wv) tot += red[wv];
  const float inv = rsqrtf(tot * (1.0f / D_) + 1e-5f);
  __bf16* orow = out + (size_t)row * D_;
#pragma unroll
  for (int i = 0; i < D_ / 256; ++i) {
    int idx = tid + i * 256;
    orow[idx] = (__bf16)(xr[idx] * inv * g[idx]);
  }
}

// ------ generic bf16 GEMM: C = A * W^T', double-buffered async LDS --------
// A: (M,K) bf16 row-major. Bt: (N,K) bf16 row-major (pre-transposed weight).
// EPI: 0 = bf16 store ; 1 = exact GELU -> bf16 ; 2 = +residual -> f32 ;
//      3 = bf16 store transposed (C[col][row], ldc = row stride of C^T)
template <int EPI>
__global__ __launch_bounds__(256)
void gemm_bf16_kernel(const __bf16* __restrict__ A, long sAz,
                      const __bf16* __restrict__ Bt, long sBz,
                      void* __restrict__ Cp, long sCz,
                      const float* __restrict__ Res,
                      int M, int N, int K, int ldc) {
  __shared__ __bf16 As[2][128][64];  // [m][k]
  __shared__ __bf16 Bs[2][64][64];   // [n][k]

  const int tid = threadIdx.x;
  const int w = tid >> 5, lane = tid & 31, hb = lane >> 4, ln = lane & 15;
  const int wm = (w & 3) * 32, wn = (w >> 2) * 32;
  const int mBase = blockIdx.y * 128, nBase = blockIdx.x * 64;
  const int z = blockIdx.z;
  const __bf16* Ag = A + (size_t)z * sAz;
  const __bf16* Bg = Bt + (size_t)z * sBz;

  f32x8 zero = {};
  f32x8 acc[2][2];
  acc[0][0] = zero; acc[0][1] = zero; acc[1][0] = zero; acc[1][1] = zero;

  const int lr = tid >> 3;        // 0..31
  const int lc = (tid & 7) * 8;   // 0..56

  auto stage = [&](int bi, int kt) {
#pragma unroll
    for (int p = 0; p < 4; ++p) {  // A tile: 128x64
      int r = p * 32 + lr;
      cp16(&Ag[(size_t)(mBase + r) * K + kt + lc], &As[bi][r][lc]);
    }
#pragma unroll
    for (int p = 0; p < 2; ++p) {  // B tile: 64 rows of W^T, straight copy
      int n = p * 32 + lr;
      cp16(&Bg[(size_t)(nBase + n) * K + kt + lc], &Bs[bi][n][lc]);
    }
  };

  stage(0, 0);
  int bi = 0;
  for (int kt = 0; kt < K; kt += 64) {
    wait_async_all();
    __syncthreads();
    if (kt + 64 < K) stage(bi ^ 1, kt + 64);  // prefetch next tile
#pragma unroll
    for (int kk = 0; kk < 64; kk += 32) {
      bf16x16 a0 = ld_afrag(&As[bi][wm + ln][kk], hb);
      bf16x16 a1 = ld_afrag(&As[bi][wm + 16 + ln][kk], hb);
      bf16x16 b0 = ld_bfrag(&Bs[bi][wn + ln][kk], hb);
      bf16x16 b1 = ld_bfrag(&Bs[bi][wn + 16 + ln][kk], hb);
      acc[0][0] = wmma_bf16(a0, b0, acc[0][0]);
      acc[0][1] = wmma_bf16(a0, b1, acc[0][1]);
      acc[1][0] = wmma_bf16(a1, b0, acc[1][0]);
      acc[1][1] = wmma_bf16(a1, b1, acc[1][1]);
    }
    bi ^= 1;
  }

#pragma unroll
  for (int i = 0; i < 2; ++i)
#pragma unroll
    for (int j = 0; j < 2; ++j)
#pragma unroll
      for (int r = 0; r < 8; ++r) {
        int row = mBase + wm + i * 16 + hb * 8 + r;
        int col = nBase + wn + j * 16 + ln;
        float v = acc[i][j][r];
        if (EPI == 1) v = 0.5f * v * (1.0f + erff(v * 0.70710678118f));
        if (EPI == 3) {
          ((__bf16*)Cp)[(size_t)z * sCz + (size_t)col * ldc + row] = (__bf16)v;
        } else if (EPI == 2) {
          size_t cidx = (size_t)z * sCz + (size_t)row * ldc + col;
          ((float*)Cp)[cidx] = v + Res[(size_t)row * ldc + col];
        } else {
          size_t cidx = (size_t)z * sCz + (size_t)row * ldc + col;
          ((__bf16*)Cp)[cidx] = (__bf16)v;
        }
      }
}

// ---------------- flash attention (causal, online softmax) ----------------
// Q/K: (H, B*S, DH) bf16. Vt: (H, DH, B*S) bf16 (pre-transposed by EPI=3).
// attn out: (B*S, D) bf16, heads concatenated.
__global__ __launch_bounds__(256)
void flash_attn_kernel(const __bf16* __restrict__ Q, const __bf16* __restrict__ Kg,
                       const __bf16* __restrict__ Vtg, __bf16* __restrict__ attn) {
  __shared__ __bf16 Ks[2][128][64];   // [key][e]
  __shared__ __bf16 Vt[2][64][128];   // [e][key]
  __shared__ __bf16 Ps[8][16][32];    // per-wave P staging

  const int tid = threadIdx.x, w = tid >> 5, lane = tid & 31;
  const int hb = lane >> 4, ln = lane & 15;
  const int qb = blockIdx.x, h = blockIdx.y, b = blockIdx.z;
  const size_t headOff  = (size_t)h * (size_t)M_ * DH_;   // Q/K head base
  const size_t headOffT = (size_t)h * (size_t)DH_ * M_;   // Vt head base
  const int mseq = qb * 128 + w * 16 + ln;  // A-frag row (seq pos)
  const size_t qrow = headOff + (size_t)(b * S_ + mseq) * DH_;

  const bf16x16 aq0 = ld_afrag(Q + qrow, hb);       // head-dim k 0..31
  const bf16x16 aq1 = ld_afrag(Q + qrow + 32, hb);  // head-dim k 32..63

  float rmax[8], rsum[8];
#pragma unroll
  for (int r = 0; r < 8; ++r) { rmax[r] = -1e30f; rsum[r] = 0.f; }
  f32x8 zero = {};
  f32x8 o[4]; o[0] = zero; o[1] = zero; o[2] = zero; o[3] = zero;

  const int lr = tid >> 3, lc = (tid & 7) * 8;
  const int ve = tid >> 4, vc = (tid & 15) * 8;

  auto stageKV = [&](int bi, int c) {
    const size_t kbase = headOff + (size_t)(b * S_ + c * 128) * DH_;
#pragma unroll
    for (int p = 0; p < 4; ++p) {
      int r = p * 32 + lr;
      cp16(&Kg[kbase + (size_t)r * DH_ + lc], &Ks[bi][r][lc]);
    }
    const size_t vbase = headOffT + (size_t)(b * S_ + c * 128);
#pragma unroll
    for (int p = 0; p < 4; ++p) {  // 64 rows (e) x 128 keys, straight copy
      int e = p * 16 + ve;
      cp16(&Vtg[vbase + (size_t)e * M_ + vc], &Vt[bi][e][vc]);
    }
  };

  const int nChunks = qb + 1;  // causal
  stageKV(0, 0);
  int bi = 0;
  for (int c = 0; c < nChunks; ++c) {
    wait_async_all();
    __syncthreads();
    if (c + 1 < nChunks) stageKV(bi ^ 1, c + 1);  // prefetch next chunk

    for (int sub = 0; sub < 4; ++sub) {  // 32 keys per subtile
      const int kb = sub * 32;
      f32x8 s0 = zero, s1 = zero;
      {
        bf16x16 b0 = ld_bfrag(&Ks[bi][kb + ln][0], hb);
        bf16x16 b1 = ld_bfrag(&Ks[bi][kb + 16 + ln][0], hb);
        s0 = wmma_bf16(aq0, b0, s0);
        s1 = wmma_bf16(aq0, b1, s1);
        b0 = ld_bfrag(&Ks[bi][kb + ln][32], hb);
        b1 = ld_bfrag(&Ks[bi][kb + 16 + ln][32], hb);
        s0 = wmma_bf16(aq1, b0, s0);
        s1 = wmma_bf16(aq1, b1, s1);
      }
      const int key0 = c * 128 + kb + ln;
      const int key1 = key0 + 16;
      float p0v[8], p1v[8], alpha[8];
#pragma unroll
      for (int r = 0; r < 8; ++r) {
        const int qpos = qb * 128 + w * 16 + hb * 8 + r;
        float v0 = s0[r] * 0.125f; const bool m0 = key0 > qpos; if (m0) v0 = -1e30f;
        float v1 = s1[r] * 0.125f; const bool m1 = key1 > qpos; if (m1) v1 = -1e30f;
        float mx = fmaxf(v0, v1);
#pragma unroll
        for (int msk = 8; msk >= 1; msk >>= 1) mx = fmaxf(mx, __shfl_xor(mx, msk, 32));
        const float nm = fmaxf(rmax[r], mx);
        alpha[r] = __expf(rmax[r] - nm);
        float e0 = m0 ? 0.f : __expf(v0 - nm);
        float e1 = m1 ? 0.f : __expf(v1 - nm);
        float ps = e0 + e1;
#pragma unroll
        for (int msk = 8; msk >= 1; msk >>= 1) ps += __shfl_xor(ps, msk, 32);
        rsum[r] = rsum[r] * alpha[r] + ps;
        rmax[r] = nm;
        p0v[r] = e0; p1v[r] = e1;
      }
#pragma unroll
      for (int f = 0; f < 4; ++f)
#pragma unroll
        for (int r = 0; r < 8; ++r) o[f][r] *= alpha[r];
      // stage P (C-layout) into LDS, reload in A-fragment layout
#pragma unroll
      for (int r = 0; r < 8; ++r) {
        Ps[w][hb * 8 + r][ln]      = (__bf16)p0v[r];
        Ps[w][hb * 8 + r][16 + ln] = (__bf16)p1v[r];
      }
      __syncthreads();
      bf16x16 pa = ld_afrag(&Ps[w][ln][0], hb);
#pragma unroll
      for (int f = 0; f < 4; ++f) {
        bf16x16 vb;
        ((bf16x8*)&vb)[0] = *(const bf16x8*)&Vt[bi][f * 16 + ln][kb + hb * 16];
        ((bf16x8*)&vb)[1] = *(const bf16x8*)&Vt[bi][f * 16 + ln][kb + hb * 16 + 8];
        o[f] = wmma_bf16(pa, vb, o[f]);
      }
      __syncthreads();
    }
    bi ^= 1;
  }

#pragma unroll
  for (int f = 0; f < 4; ++f)
#pragma unroll
    for (int r = 0; r < 8; ++r) {
      int ms = qb * 128 + w * 16 + hb * 8 + r;
      float val = o[f][r] / rsum[r];
      size_t idx = (size_t)(b * S_ + ms) * D_ + h * DH_ + f * 16 + ln;
      attn[idx] = (__bf16)val;
    }
}

// ---------------- host-side orchestration ----------------
extern "C" void kernel_launch(void* const* d_in, const int* in_sizes, int n_in,
                              void* d_out, int out_size, void* d_ws, size_t ws_size,
                              hipStream_t stream) {
  (void)in_sizes; (void)n_in; (void)out_size; (void)ws_size;
  const float* x  = (const float*)d_in[0];
  const float* Wq = (const float*)d_in[1];
  const float* Wk = (const float*)d_in[2];
  const float* Wv = (const float*)d_in[3];
  const float* Wo = (const float*)d_in[4];
  const float* w1 = (const float*)d_in[5];
  const float* w2 = (const float*)d_in[6];
  const float* g1 = (const float*)d_in[7];
  const float* g2 = (const float*)d_in[8];

  char* p = (char*)d_ws;
  auto alloc = [&](size_t bytes) {
    char* r = p;
    p += (bytes + 255) & ~(size_t)255;
    return r;
  };
  // all weights stored pre-transposed (N,K) in bf16
  __bf16* WqT = (__bf16*)alloc((size_t)H_ * D_ * DH_ * 2);
  __bf16* WkT = (__bf16*)alloc((size_t)H_ * D_ * DH_ * 2);
  __bf16* WvT = (__bf16*)alloc((size_t)H_ * D_ * DH_ * 2);
  __bf16* WoT = (__bf16*)alloc((size_t)D_ * D_ * 2);
  __bf16* w1T = (__bf16*)alloc((size_t)D_ * DFF_ * 2);
  __bf16* w2T = (__bf16*)alloc((size_t)DFF_ * D_ * 2);
  __bf16* h1   = (__bf16*)alloc((size_t)M_ * D_ * 2);
  __bf16* qb_  = (__bf16*)alloc((size_t)H_ * M_ * DH_ * 2);
  __bf16* kb_  = (__bf16*)alloc((size_t)H_ * M_ * DH_ * 2);
  __bf16* vt_  = (__bf16*)alloc((size_t)H_ * DH_ * M_ * 2);  // V transposed
  __bf16* attn = (__bf16*)alloc((size_t)M_ * D_ * 2);
  float*  x1   = (float*)alloc((size_t)M_ * D_ * 4);
  __bf16* h2   = (__bf16*)alloc((size_t)M_ * D_ * 2);
  __bf16* ffn1 = (__bf16*)alloc((size_t)M_ * DFF_ * 2);

  const dim3 blk(256);

  // weight transpose + downconvert: [z][K][N] f32 -> [z][N][K] bf16
  cvt_transpose<<<dim3(DH_ / 32, D_ / 32, H_), blk, 0, stream>>>(Wq, WqT, D_, DH_);
  cvt_transpose<<<dim3(DH_ / 32, D_ / 32, H_), blk, 0, stream>>>(Wk, WkT, D_, DH_);
  cvt_transpose<<<dim3(DH_ / 32, D_ / 32, H_), blk, 0, stream>>>(Wv, WvT, D_, DH_);
  cvt_transpose<<<dim3(D_ / 32, D_ / 32, 1), blk, 0, stream>>>(Wo, WoT, D_, D_);
  cvt_transpose<<<dim3(DFF_ / 32, D_ / 32, 1), blk, 0, stream>>>(w1, w1T, D_, DFF_);
  cvt_transpose<<<dim3(D_ / 32, DFF_ / 32, 1), blk, 0, stream>>>(w2, w2T, DFF_, D_);

  // h1 = rmsnorm(x, g1)
  rmsnorm_bf16<<<M_, blk, 0, stream>>>(x, g1, h1);

  // q/k = h1 @ Wq/Wk[h] -> (H, M, DH) bf16 ; v stored transposed (H, DH, M)
  gemm_bf16_kernel<0><<<dim3(1, M_ / 128, H_), blk, 0, stream>>>(
      h1, 0, WqT, (long)D_ * DH_, qb_, (long)M_ * DH_, nullptr, M_, DH_, D_, DH_);
  gemm_bf16_kernel<0><<<dim3(1, M_ / 128, H_), blk, 0, stream>>>(
      h1, 0, WkT, (long)D_ * DH_, kb_, (long)M_ * DH_, nullptr, M_, DH_, D_, DH_);
  gemm_bf16_kernel<3><<<dim3(1, M_ / 128, H_), blk, 0, stream>>>(
      h1, 0, WvT, (long)D_ * DH_, vt_, (long)DH_ * M_, nullptr, M_, DH_, D_, M_);

  // causal flash attention -> attn (M, D) bf16
  flash_attn_kernel<<<dim3(S_ / 128, H_, B_), blk, 0, stream>>>(qb_, kb_, vt_, attn);

  // x1 = x + attn @ Wo  (f32)
  gemm_bf16_kernel<2><<<dim3(D_ / 64, M_ / 128, 1), blk, 0, stream>>>(
      attn, 0, WoT, 0, x1, 0, x, M_, D_, D_, D_);

  // h2 = rmsnorm(x1, g2)
  rmsnorm_bf16<<<M_, blk, 0, stream>>>(x1, g2, h2);

  // ffn1 = gelu(h2 @ w1) bf16
  gemm_bf16_kernel<1><<<dim3(DFF_ / 64, M_ / 128, 1), blk, 0, stream>>>(
      h2, 0, w1T, 0, ffn1, 0, nullptr, M_, DFF_, D_, DFF_);

  // out = x1 + ffn1 @ w2 (f32)
  gemm_bf16_kernel<2><<<dim3(D_ / 64, M_ / 128, 1), blk, 0, stream>>>(
      ffn1, 0, w2T, 0, d_out, 0, x1, M_, D_, DFF_, D_);
}
